// MultiHeadAttention_32633161515503
// MI455X (gfx1250) — matrix operand order repaired and checked by reading, not compiled, not run on hardware
//
#include <hip/hip_runtime.h>

typedef __bf16 bf16;
typedef __attribute__((ext_vector_type(16))) __bf16 v16bf;
typedef __attribute__((ext_vector_type(8)))  __bf16 v8bf;
typedef __attribute__((ext_vector_type(4)))  __bf16 v4bf;
typedef __attribute__((ext_vector_type(8)))  float  v8f;

#define B_  2
#define S_  2048
#define D_  1024
#define H_  16
#define DK_ 64
#define M_  (B_*S_)

// ---------------- CDNA5 async copy (ASYNCcnt path) -------------------------
// global_load_async_to_lds_b128: per-lane 16B global -> LDS, no VGPR round
// trip, tracked with ASYNCcnt (not LOADcnt/DScnt).

static __device__ __forceinline__ unsigned ldsoff(const void* p) {
  // generic pointers to LDS carry the LDS byte address in their low 32 bits
  return (unsigned)(size_t)p;
}

static __device__ __forceinline__ void async_ld16(unsigned lds, const bf16* g) {
  asm volatile("global_load_async_to_lds_b128 %0, %1, off"
               :: "v"(lds), "v"(g) : "memory");
}

static __device__ __forceinline__ void wait_async0() {
  asm volatile("s_wait_asynccnt 0x0" ::: "memory");
}

// ---------------- fragment helpers (CDNA5 16-bit WMMA layouts) -------------

static __device__ __forceinline__ int laneid() { return (int)(threadIdx.x & 31u); }

static __device__ __forceinline__ v16bf cat8(v8bf a, v8bf b) {
  return __builtin_shufflevector(a, b, 0,1,2,3,4,5,6,7,8,9,10,11,12,13,14,15);
}

// A-matrix 16x32 bf16: lane = M (0..15), half-wave selects K-subgroups.
// lanes 0-15: K = {0..7, 16..23}; lanes 16-31: K = {8..15, 24..31}
static __device__ __forceinline__ v16bf frag_a(const bf16* base, int ld) {
  int l = laneid();
  const bf16* p = base + (l & 15) * ld + 8 * (l >> 4);
  v8bf c0 = *(const v8bf*)p;
  v8bf c1 = *(const v8bf*)(p + 16);
  return cat8(c0, c1);
}

// B-matrix 32x16 bf16, memory holds rows of length `ld` along K:
// lane = N (0..15), lanes 0-15 hold K={0..7,16..23}, lanes 16-31 hold K={8..15,24..31}
static __device__ __forceinline__ v16bf frag_b(const bf16* base, int ld) {
  int l = laneid();
  const bf16* p = base + (long)(l & 15) * ld + 8 * (l >> 4);
  v8bf c0 = *(const v8bf*)p;
  v8bf c1 = *(const v8bf*)(p + 16);
  return cat8(c0, c1);
}

static __device__ __forceinline__ v8f wmma_bf16(v16bf a, v16bf b, v8f c) {
  // (neg_a, A, neg_b, B, c_mod, C, reuse_a, reuse_b)
  return __builtin_amdgcn_wmma_f32_16x16x32_bf16(false, a, false, b, (short)0, c,
                                                 false, false);
}

// ---------------- f32 -> bf16 conversion ----------------------------------

__global__ void cvt_f32_bf16(const float* __restrict__ in, bf16* __restrict__ out,
                             int n) {
  int i = ((int)blockIdx.x * 256 + (int)threadIdx.x) * 4;
  if (i >= n) return;
  float4 f = *(const float4*)(in + i);
  v4bf o;
  o[0] = (bf16)f.x; o[1] = (bf16)f.y; o[2] = (bf16)f.z; o[3] = (bf16)f.w;
  *(v4bf*)(out + i) = o;
}

// ---------------- tiled bf16 GEMM:  C = A @ W^T + bias --------------------
// Double-buffered LDS pipeline fed by ASYNC global->LDS copies:
//   wait_asynccnt(0) ; barrier ; issue async for tile i+1 ; 4 WMMAs on tile i
// mode 0: bf16 out, head-split  [b,h,s,dk]
// mode 1: f32  out, flat        [m,n]
// mode 2: bf16 out, head-split transposed [b,h,dk,s]

#define GBM 128
#define GBN 64
#define GKB 32
#define KP  40   // padded K stride in LDS (bank-conflict avoidance, 16B aligned)

__global__ __launch_bounds__(256) void gemm_bf16(
    const bf16* __restrict__ A, const bf16* __restrict__ W,
    const float* __restrict__ bias, void* __restrict__ outp,
    int M, int N, int K, int mode) {
  __shared__ bf16 sA[2][GBM * KP];
  __shared__ bf16 sW[2][GBN * KP];
  int tid = threadIdx.x;
  int wave = tid >> 5, lane = tid & 31;
  int m0 = blockIdx.y * GBM, n0 = blockIdx.x * GBN;
  int wr = (wave >> 1) * 32, wc = (wave & 1) * 32;

  // async staging: 16B chunks; A tile 128x32 = 512 chunks (2/thread),
  // W tile 64x32 = 256 chunks (1/thread)
  int crow = tid >> 2;              // 0..63
  int coff = (tid & 3) * 8;         // 0,8,16,24 (elements)

  auto issue = [&](int buf, int k0) {
    const bf16* a0g = A + (long)(m0 + crow) * K + k0 + coff;
    const bf16* a1g = A + (long)(m0 + 64 + crow) * K + k0 + coff;
    const bf16* wg  = W + (long)(n0 + crow) * K + k0 + coff;
    async_ld16(ldsoff(&sA[buf][crow * KP + coff]), a0g);
    async_ld16(ldsoff(&sA[buf][(64 + crow) * KP + coff]), a1g);
    async_ld16(ldsoff(&sW[buf][crow * KP + coff]), wg);
  };

  v8f acc[2][2] = {};
  int nsteps = K / GKB;
  issue(0, 0);

  for (int i = 0; i < nsteps; ++i) {
    int cur = i & 1;
    wait_async0();        // my async writes into buf[cur] done
    __syncthreads();      // everyone's writes visible; prior reads of buf[cur^1] done
    if (i + 1 < nsteps) issue(cur ^ 1, (i + 1) * GKB);

    v16bf a0 = frag_a(&sA[cur][(wr +  0) * KP], KP);
    v16bf a1 = frag_a(&sA[cur][(wr + 16) * KP], KP);
    v16bf b0 = frag_b(&sW[cur][(wc +  0) * KP], KP);
    v16bf b1 = frag_b(&sW[cur][(wc + 16) * KP], KP);
    acc[0][0] = wmma_bf16(a0, b0, acc[0][0]);
    acc[0][1] = wmma_bf16(a0, b1, acc[0][1]);
    acc[1][0] = wmma_bf16(a1, b0, acc[1][0]);
    acc[1][1] = wmma_bf16(a1, b1, acc[1][1]);
  }

  // epilogue: C frag layout -> VGPR r: lanes0-15 M=r, lanes16-31 M=r+8
  for (int tm = 0; tm < 2; ++tm)
    for (int tn = 0; tn < 2; ++tn)
      for (int r = 0; r < 8; ++r) {
        int ml = wr + tm * 16 + r + 8 * (lane >> 4);
        int nl = wc + tn * 16 + (lane & 15);
        int mg = m0 + ml, ng = n0 + nl;
        float v = acc[tm][tn][r] + bias[ng];
        if (mode == 1) {
          ((float*)outp)[(long)mg * N + ng] = v;
        } else {
          int bb = mg >> 11, s = mg & (S_ - 1);
          int hh = ng >> 6,  dc = ng & (DK_ - 1);
          bf16* o = (bf16*)outp;
          if (mode == 0)
            o[(((long)(bb * H_ + hh)) * S_ + s) * DK_ + dc] = (bf16)v;
          else
            o[(((long)(bb * H_ + hh)) * DK_ + dc) * S_ + s] = (bf16)v;
        }
      }
}

// ---------------- fused attention: scores -> softmax -> attn@V ------------
// One workgroup = one (b, h, 32-row q block). The full 32 x 2048 score
// stripe lives in LDS (~132 KB of the 320 KB WGP LDS), so scores touch HBM
// exactly once (the mandatory attn output write).

#define QB 32
#define SP (S_ + 8)     // padded score row stride (bf16, keeps 16B alignment)
#define QP (DK_ + 8)    // padded Q row stride

__global__ __launch_bounds__(256) void attn_fused(
    const bf16* __restrict__ qh, const bf16* __restrict__ kh,
    const bf16* __restrict__ vhT, float* __restrict__ attn,
    bf16* __restrict__ outh) {
  __shared__ bf16 sS[QB * SP];
  __shared__ bf16 sQ[QB * QP];

  int tid = threadIdx.x, wave = tid >> 5, lane = tid & 31;
  int qblk = blockIdx.x, h = blockIdx.y, b = blockIdx.z;
  int q0 = qblk * QB, qmax = q0 + QB - 1;

  const bf16* qhb = qh  + ((long)(b * H_ + h) * S_) * DK_;
  const bf16* khb = kh  + ((long)(b * H_ + h) * S_) * DK_;
  const bf16* vtb = vhT + ((long)(b * H_ + h) * DK_) * S_;

  // stage Q tile (32x64) via one async B128 per thread
  {
    int row = tid >> 3, off = (tid & 7) * 8;
    async_ld16(ldsoff(&sQ[row * QP + off]),
               qhb + (long)(q0 + row) * DK_ + off);
  }
  // init score stripe to -inf-ish (covers causally-skipped tiles)
  for (int i = tid; i < QB * SP; i += 256) sS[i] = (bf16)(-1e30f);
  wait_async0();
  __syncthreads();

  // phase 1: scores = (Q K^T) * 1/sqrt(dk), causal-masked, bf16 into LDS
  for (int nt = wave; nt * 16 <= qmax && nt < S_ / 16; nt += 8) {
    int kb = nt * 16;
    if ((nt + 8) * 16 <= qmax)   // near-scope prefetch of next K tile
      __builtin_prefetch((const void*)(khb + (long)(kb + 128) * DK_), 0, 3);
    for (int tm = 0; tm < 2; ++tm) {
      v8f acc = {};
      for (int ks = 0; ks < 2; ++ks) {            // dk = 64 -> 2 k-steps
        v16bf a  = frag_a(&sQ[(tm * 16) * QP + ks * 32], QP);
        v16bf bb = frag_b(khb + (long)kb * DK_ + ks * 32, DK_);
        acc = wmma_bf16(a, bb, acc);
      }
      int n = lane & 15, hw = lane >> 4;
      for (int r = 0; r < 8; ++r) {
        int ml = tm * 16 + r + 8 * hw;
        int ka = kb + n;
        float v = acc[r] * 0.125f;                 // 1/sqrt(64)
        if (ka > q0 + ml) v = -1e30f;              // causal mask
        sS[ml * SP + kb + n] = (bf16)v;
      }
    }
  }
  __syncthreads();

  // phase 2: row softmax (4 rows per wave), write normalized attn to HBM
  for (int rr = 0; rr < 4; ++rr) {
    int ri = wave * 4 + rr;
    float mx = -1e30f;
    for (int j = lane; j < S_; j += 32) mx = fmaxf(mx, (float)sS[ri * SP + j]);
    for (int off = 16; off >= 1; off >>= 1) mx = fmaxf(mx, __shfl_xor(mx, off, 32));
    float sum = 0.f;
    for (int j = lane; j < S_; j += 32) {
      float p = __expf((float)sS[ri * SP + j] - mx);
      sS[ri * SP + j] = (bf16)p;
      sum += p;
    }
    for (int off = 16; off >= 1; off >>= 1) sum += __shfl_xor(sum, off, 32);
    float inv = 1.f / sum;
    float* ao = attn + ((long)((b * H_ + h) * S_ + (q0 + ri))) * S_;
    for (int j = lane; j < S_; j += 32) {
      float pv = (float)sS[ri * SP + j] * inv;
      sS[ri * SP + j] = (bf16)pv;
      ao[j] = pv;                                  // single coalesced f32 write
    }
  }
  __syncthreads();

  // phase 3: out = attn @ V (K-loop stops at the causal boundary)
  int tm = wave >> 2, tn = wave & 3;               // 2 x 4 output tiles of 16x16
  v8f acc = {};
  int ksEnd = q0 / 32 + 1;
  for (int ks = 0; ks < ksEnd; ++ks) {
    v16bf a  = frag_a(&sS[(tm * 16) * SP + ks * 32], SP);
    v16bf bb = frag_b(vtb + (long)(tn * 16) * S_ + ks * 32, S_);
    acc = wmma_bf16(a, bb, acc);
  }
  int n = lane & 15, hw = lane >> 4;
  for (int r = 0; r < 8; ++r) {
    int ml = tm * 16 + r + 8 * hw;
    int s = q0 + ml;
    int dcol = h * DK_ + tn * 16 + n;
    outh[((long)(b * S_ + s)) * D_ + dcol] = (bf16)acc[r];
  }
}

// ---------------- host side ------------------------------------------------

extern "C" void kernel_launch(void* const* d_in, const int* in_sizes, int n_in,
                              void* d_out, int out_size, void* d_ws, size_t ws_size,
                              hipStream_t stream) {
  const float* q  = (const float*)d_in[0];
  const float* k  = (const float*)d_in[1];
  const float* v  = (const float*)d_in[2];
  // d_in[3] = mask (causal tril, hardcoded), d_in[4] = num_heads (=16)
  const float* Wq = (const float*)d_in[5];
  const float* bq = (const float*)d_in[6];
  const float* Wk = (const float*)d_in[7];
  const float* bk = (const float*)d_in[8];
  const float* Wv = (const float*)d_in[9];
  const float* bv = (const float*)d_in[10];
  const float* Wo = (const float*)d_in[11];
  const float* bo = (const float*)d_in[12];
  (void)in_sizes; (void)n_in; (void)out_size; (void)ws_size;

  char* ws = (char*)d_ws;
  const size_t MB = 1024 * 1024;
  bf16* Wqb = (bf16*)(ws +  0 * MB);   // 4 weight matrices, 2 MB each
  bf16* Wkb = (bf16*)(ws +  2 * MB);
  bf16* Wvb = (bf16*)(ws +  4 * MB);
  bf16* Wob = (bf16*)(ws +  6 * MB);
  bf16* Xq  = (bf16*)(ws +  8 * MB);   // bf16 activations, 8 MB each
  bf16* Xk  = (bf16*)(ws + 16 * MB);
  bf16* Xv  = (bf16*)(ws + 24 * MB);
  bf16* Qh  = (bf16*)(ws + 32 * MB);   // head-split Q/K, V transposed
  bf16* Kh  = (bf16*)(ws + 40 * MB);
  bf16* VhT = (bf16*)(ws + 48 * MB);
  bf16* Oh  = (bf16*)(ws + 56 * MB);   // merged-head attention output (bf16)

  const int ND = M_ * D_;              // 4M elements
  const int NW = D_ * D_;              // 1M elements
  cvt_f32_bf16<<<ND / 1024, 256, 0, stream>>>(q,  Xq,  ND);
  cvt_f32_bf16<<<ND / 1024, 256, 0, stream>>>(k,  Xk,  ND);
  cvt_f32_bf16<<<ND / 1024, 256, 0, stream>>>(v,  Xv,  ND);
  cvt_f32_bf16<<<NW / 1024, 256, 0, stream>>>(Wq, Wqb, NW);
  cvt_f32_bf16<<<NW / 1024, 256, 0, stream>>>(Wk, Wkb, NW);
  cvt_f32_bf16<<<NW / 1024, 256, 0, stream>>>(Wv, Wvb, NW);
  cvt_f32_bf16<<<NW / 1024, 256, 0, stream>>>(Wo, Wob, NW);

  dim3 gg(D_ / GBN, M_ / GBM);         // (16, 32)
  gemm_bf16<<<gg, 256, 0, stream>>>(Xq, Wqb, bq, Qh,  M_, D_, D_, 0);
  gemm_bf16<<<gg, 256, 0, stream>>>(Xk, Wkb, bk, Kh,  M_, D_, D_, 0);
  gemm_bf16<<<gg, 256, 0, stream>>>(Xv, Wvb, bv, VhT, M_, D_, D_, 2);

  float* out  = (float*)d_out;
  float* attn = out + (size_t)M_ * D_;
  attn_fused<<<dim3(S_ / QB, H_, B_), 256, 0, stream>>>(Qh, Kh, VhT, attn, Oh);

  gemm_bf16<<<gg, 256, 0, stream>>>(Oh, Wob, bo, out, M_, D_, D_, 1);
}
